// HybridSpatialOutbreakSimulator_13597866459496
// MI455X (gfx1250) — matrix operand order repaired and checked by reading, not compiled, not run on hardware
//
#include <hip/hip_runtime.h>
#include <hip/hip_bf16.h>

// ---------------------------------------------------------------------------
// HybridSpatialOutbreakSimulator — CDNA5 (gfx1250) implementation.
// Edge MLPs (92% of FLOPs) use v_wmma_f32_16x16x32_f16 with B-fragments
// pre-swizzled in LDS (ds_load_b128 feeds); segment softmax via coded-uint
// atomicMax + float atomicAdd; node GRU phase is scalar VALU with
// wave-uniform (SMEM) weight loads.
// ---------------------------------------------------------------------------

#define TT 8
#define NN 10000
#define FF 16
#define EE 320000
#define HH 64
#define LDIM 3
#define ENVD 5
#define C_TEMP 11
#define C_SAL  12
#define C_UI   13
#define C_VI   14

typedef __attribute__((ext_vector_type(16))) _Float16 v16h;
typedef __attribute__((ext_vector_type(8)))  float    v8f;
typedef __attribute__((ext_vector_type(4)))  unsigned uint4v;
typedef __attribute__((ext_vector_type(8)))  unsigned uint8v;
typedef __attribute__((ext_vector_type(4)))  float    float4v;

// ------------------------------ helpers ------------------------------------

__device__ __forceinline__ float softplusf(float x) {
    return (x > 20.f) ? x : log1pf(__expf(x));
}

__device__ __forceinline__ unsigned packh(float a, float b) {
    union { _Float16 h[2]; unsigned u; } x;
    x.h[0] = (_Float16)a; x.h[1] = (_Float16)b;
    return x.u;
}

// A fragment: 16x32 f16, row-major source in LDS with leading dim lda (halfs).
// ISA 7.12.2 layout: lane row = lane&15, half = lane>>4; the 16 halves a lane
// needs are K = 8*half..8*half+7 and K = 16+8*half..16+8*half+7 -> two
// contiguous 16-byte runs -> 2x ds_load_b128.
__device__ __forceinline__ v16h load_a_frag(const _Float16* A, int lda, int lane) {
    const int row  = lane & 15;
    const int half = lane >> 4;
    const _Float16* base = A + row * lda + 8 * half;
    const uint4v lo = *(const uint4v*)(const void*)(base);        // K = 8h .. 8h+7
    const uint4v hi = *(const uint4v*)(const void*)(base + 16);   // K = 16+8h .. 16+8h+7
    uint8v u;
#pragma unroll
    for (int i = 0; i < 4; ++i) { u[i] = lo[i]; u[4 + i] = hi[i]; }
    return __builtin_bit_cast(v16h, u);
}

// B fragment from pre-swizzled LDS buffer: frag stores lane-major 16 halves
// (order: this lane's 16 K values at n-offset lane&15) -> 2x ds_load_b128.
__device__ __forceinline__ v16h load_b_frag_sw(const _Float16* frag, int lane) {
    const _Float16* p = frag + lane * 16;
    const uint4v lo = *(const uint4v*)(const void*)(p);
    const uint4v hi = *(const uint4v*)(const void*)(p + 8);
    uint8v u;
#pragma unroll
    for (int i = 0; i < 4; ++i) { u[i] = lo[i]; u[4 + i] = hi[i]; }
    return __builtin_bit_cast(v16h, u);
}

// Stage W [outDim][kdim] (row-major f32) into B-fragment-swizzled f16 LDS.
// Frag f = nt*nFragK + kc holds 512 halves: index r = lane*16 + t where
// n-offset = lane&15 and K = 8*(lane>>4) + t (+8 if t>=8) (table order).
__device__ __forceinline__ void stage_bfrags(_Float16* dst, const float* W,
                                             int ldw, int kdim,
                                             int nFragN, int nFragK, int tid) {
    const int total = nFragN * nFragK * 512;
    for (int i = tid; i < total; i += 64) {
        const int f    = i >> 9;
        const int r    = i & 511;
        const int lane = r >> 4;
        const int t    = r & 15;
        const int nt   = f / nFragK;
        const int kc   = f - nt * nFragK;
        const int k    = kc * 32 + 8 * (lane >> 4) + t + ((t >= 8) ? 8 : 0);
        const int n    = nt * 16 + (lane & 15);
        dst[i] = (_Float16)((k < kdim) ? W[n * ldw + k] : 0.f);
    }
}

// ------------------------------ LDS layout ---------------------------------

struct __align__(16) WaveLDS {
    _Float16 A1[16 * 32];    // edge attrs, K padded to 32
    _Float16 Y1[16 * 64];    // relu(attr @ ee_w1^T + b1)
    _Float16 X2[16 * 128];   // [EF | h_src]
    float    Y3[16 * 64];    // relu(X2 @ at_w1^T + b1)
};

struct __align__(16) BlockLDS {
    _Float16 bf_ee1[4  * 512];   // GEMM1 B frags (K padded 4->32)
    _Float16 bf_ee2[8  * 512];   // GEMM2 B frags (nt*2+kc)
    _Float16 bf_at1[16 * 512];   // GEMM3 B frags (nt*4+kc)
    float    at_w2f[64];
    float    ee_b1f[64];
    float    ee_b2f[64];
    float    at_b1f[64];
    WaveLDS  w[2];
};  // ~51 KB static LDS

// ------------------------- kernel 0: zero fill -----------------------------

__global__ void zero_f_kernel(float* __restrict__ p, int n) {
    int i = blockIdx.x * blockDim.x + threadIdx.x;
    if (i < n) p[i] = 0.f;
}

__global__ void init_step_kernel(float* __restrict__ pressure,
                                 unsigned* __restrict__ amax_u,
                                 float* __restrict__ asum) {
    int i = blockIdx.x * blockDim.x + threadIdx.x;
    if (i < NN * HH) pressure[i] = 0.f;
    if (i < NN) { amax_u[i] = 0u; asum[i] = 0.f; }
}

// -------------------- kernel 1: edge MLPs + logits (WMMA) ------------------
// blockDim = 64 (2 waves). Each wave processes 16-edge tiles.

__global__ void __launch_bounds__(64)
edge_logits_kernel(const float* __restrict__ xt, const float* __restrict__ hbuf,
                   const float* __restrict__ edist, const float* __restrict__ edir,
                   const int* __restrict__ eidx,
                   const float* __restrict__ ee_w1, const float* __restrict__ ee_b1,
                   const float* __restrict__ ee_w2, const float* __restrict__ ee_b2,
                   const float* __restrict__ at_w1, const float* __restrict__ at_b1,
                   const float* __restrict__ at_w2, const float* __restrict__ at_b2,
                   float* __restrict__ logits, unsigned* __restrict__ amax_u,
                   int nTiles) {
    __shared__ BlockLDS L;

    const int tid  = threadIdx.x;
    const int lane = tid & 31;
    const int wv   = tid >> 5;

    // ---- stage weights pre-swizzled into B-fragment layout (once/block) ----
    stage_bfrags(L.bf_ee1, ee_w1,   4,   4, 4, 1, tid);
    stage_bfrags(L.bf_ee2, ee_w2,  64,  64, 4, 2, tid);
    stage_bfrags(L.bf_at1, at_w1, 128, 128, 4, 4, tid);
    // tid always < 64
    L.at_w2f[tid] = at_w2[tid];
    L.ee_b1f[tid] = ee_b1[tid];
    L.ee_b2f[tid] = ee_b2[tid];
    L.at_b1f[tid] = at_b1[tid];
    __syncthreads();

    const float at_b2v = at_b2[0];
    WaveLDS& W = L.w[wv];

    const int waveId     = blockIdx.x * 2 + wv;
    const int totalWaves = gridDim.x * 2;

    for (int tile = waveId; tile < nTiles; tile += totalWaves) {
        const int e0 = tile * 16;

        // ---- gather: edge attrs (lanes 0..15) + h_src f16 (all lanes) ----
        float wflux_keep = 0.f;
        int   myDst      = 0;
        if (lane < 16) {
            const int e = e0 + lane;
            const int s = eidx[e];
            myDst       = eidx[EE + e];
            const float dist = edist[e];
            const float dirx = edir[2 * e], diry = edir[2 * e + 1];
            const float* xs = xt + (size_t)s * FF;
            const float flux = xs[C_UI] * dirx + xs[C_VI] * diry;
            const float wfl  = fmaxf(flux, 0.f) * __expf(-dist * (1.f / 15.f));
            wflux_keep = wfl;
            // packed row write: 2 data dwords + 14 zero dwords
            unsigned* r = (unsigned*)(void*)&W.A1[lane * 32];
            r[0] = packh(dist, wfl);
            r[1] = packh(xs[C_TEMP], xs[C_SAL]);
#pragma unroll
            for (int k = 2; k < 16; ++k) r[k] = 0u;
        }
        {
            const int row = lane >> 1;
            const int s   = eidx[e0 + row];
            const float4v* hs =
                (const float4v*)(const void*)(hbuf + (size_t)s * HH + (lane & 1) * 32);
            uint4v* d =
                (uint4v*)(void*)&W.X2[row * 128 + 64 + (lane & 1) * 32];
#pragma unroll
            for (int q = 0; q < 4; ++q) {
                const float4v v0 = hs[2 * q];
                const float4v v1 = hs[2 * q + 1];
                uint4v pk;
                pk[0] = packh(v0[0], v0[1]);
                pk[1] = packh(v0[2], v0[3]);
                pk[2] = packh(v1[0], v1[1]);
                pk[3] = packh(v1[2], v1[3]);
                d[q] = pk;
            }
        }
        __syncthreads();

        // ---- GEMM1: Y1 = relu(A1(16x4,pad32) @ ee_w1^T + b1) ----
        {
            const v16h a = load_a_frag(W.A1, 32, lane);
#pragma unroll
            for (int nt = 0; nt < 4; ++nt) {
                const v16h b = load_b_frag_sw(L.bf_ee1 + nt * 512, lane);
                const float bias = L.ee_b1f[nt * 16 + (lane & 15)];
                v8f c;
#pragma unroll
                for (int i = 0; i < 8; ++i) c[i] = bias;
                c = __builtin_amdgcn_wmma_f32_16x16x32_f16(false, a, false, b,
                                                           (short)0, c, false, false);
                const int col = nt * 16 + (lane & 15);
#pragma unroll
                for (int i = 0; i < 8; ++i) {
                    const int m = i + 8 * (lane >> 4);
                    W.Y1[m * 64 + col] = (_Float16)fmaxf(c[i], 0.f);
                }
            }
        }
        __syncthreads();

        // ---- GEMM2: EF = Y1(16x64) @ ee_w2^T + b2  (no relu) -> X2[:, :64] ----
#pragma unroll
        for (int nt = 0; nt < 4; ++nt) {
            const float bias = L.ee_b2f[nt * 16 + (lane & 15)];
            v8f c;
#pragma unroll
            for (int i = 0; i < 8; ++i) c[i] = bias;
#pragma unroll
            for (int kc = 0; kc < 2; ++kc) {
                const v16h a = load_a_frag(W.Y1 + kc * 32, 64, lane);
                const v16h b = load_b_frag_sw(L.bf_ee2 + (nt * 2 + kc) * 512, lane);
                c = __builtin_amdgcn_wmma_f32_16x16x32_f16(false, a, false, b,
                                                           (short)0, c, false, false);
            }
            const int col = nt * 16 + (lane & 15);
#pragma unroll
            for (int i = 0; i < 8; ++i) {
                const int m = i + 8 * (lane >> 4);
                W.X2[m * 128 + col] = (_Float16)c[i];
            }
        }
        __syncthreads();

        // ---- GEMM3: Y3 = relu(X2(16x128) @ at_w1^T + b1) ----
#pragma unroll
        for (int nt = 0; nt < 4; ++nt) {
            const float bias = L.at_b1f[nt * 16 + (lane & 15)];
            v8f c;
#pragma unroll
            for (int i = 0; i < 8; ++i) c[i] = bias;
#pragma unroll
            for (int kc = 0; kc < 4; ++kc) {
                const v16h a = load_a_frag(W.X2 + kc * 32, 128, lane);
                const v16h b = load_b_frag_sw(L.bf_at1 + (nt * 4 + kc) * 512, lane);
                c = __builtin_amdgcn_wmma_f32_16x16x32_f16(false, a, false, b,
                                                           (short)0, c, false, false);
            }
            const int col = nt * 16 + (lane & 15);
#pragma unroll
            for (int i = 0; i < 8; ++i) {
                const int m = i + 8 * (lane >> 4);
                W.Y3[m * 64 + col] = fmaxf(c[i], 0.f);
            }
        }
        __syncthreads();

        // ---- logits: Y3 @ at_w2^T + b2 + log(wflux+1e-8); segment max ----
        if (lane < 16) {
            float acc = at_b2v;
            const float* y = &W.Y3[lane * 64];
#pragma unroll
            for (int k = 0; k < 64; ++k) acc += y[k] * L.at_w2f[k];
            const float logit = acc + __logf(wflux_keep + 1e-8f);
            logits[e0 + lane] = logit;
            unsigned bits = __float_as_uint(logit);
            unsigned code = (bits & 0x80000000u) ? ~bits : (bits | 0x80000000u);
            atomicMax(&amax_u[myDst], code);
        }
        __syncthreads();
    }
}

// ----------------- kernel 2: exp(logit - amax[dst]) + segsum ---------------

__global__ void edge_softmax_kernel(const float* __restrict__ logits,
                                    const int* __restrict__ eidx,
                                    const unsigned* __restrict__ amax_u,
                                    float* __restrict__ aexp,
                                    float* __restrict__ asum) {
    int e = blockIdx.x * blockDim.x + threadIdx.x;
    if (e >= EE) return;
    const int d = eidx[EE + e];
    const unsigned code = amax_u[d];
    const unsigned bits = (code & 0x80000000u) ? (code & 0x7FFFFFFFu) : ~code;
    const float amax = __uint_as_float(bits);
    const float ae = __expf(logits[e] - amax);
    aexp[e] = ae;
    atomicAdd(&asum[d], ae);
}

// ----------------- kernel 3: pressure scatter-add --------------------------
// one thread per (edge, 16 columns of H)

__global__ void pressure_kernel(const float* __restrict__ aexp,
                                const float* __restrict__ asum,
                                const float* __restrict__ lice_cur,
                                const float* __restrict__ hbuf,
                                const int* __restrict__ eidx,
                                const float* __restrict__ log_beta,
                                float* __restrict__ pressure) {
    int idx = blockIdx.x * blockDim.x + threadIdx.x;
    if (idx >= EE * 4) return;
    const int e = idx >> 2;
    const int part = idx & 3;
    const int s = eidx[e];
    const int d = eidx[EE + e];
    const float beta = __expf(log_beta[0]);
    const float aw = aexp[e] / (asum[d] + 1e-8f);
    const float coeff = aw * lice_cur[(size_t)s * LDIM] * beta;
    const float* hs = hbuf + (size_t)s * HH + part * 16;
    float* pd = pressure + (size_t)d * HH + part * 16;
#pragma unroll
    for (int k = 0; k < 16; ++k) atomicAdd(&pd[k], coeff * hs[k]);
}

// ----------------- kernel 4: node update (env/GRU/decoder) -----------------
// one thread per node; weight addresses are wave-uniform -> scalar loads.

__global__ void node_update_kernel(const float* __restrict__ xt,
                                   const float* __restrict__ hbuf,
                                   const float* __restrict__ pressure,
                                   const float* __restrict__ lice_cur,
                                   const float* __restrict__ en_w1, const float* __restrict__ en_b1,
                                   const float* __restrict__ en_w2, const float* __restrict__ en_b2,
                                   const float* __restrict__ fe_w1, const float* __restrict__ fe_b1,
                                   const float* __restrict__ fe_w2, const float* __restrict__ fe_b2,
                                   const float* __restrict__ fe_w3, const float* __restrict__ fe_b3,
                                   const float* __restrict__ wih, const float* __restrict__ whh,
                                   const float* __restrict__ bih, const float* __restrict__ bhh,
                                   const float* __restrict__ de_w1, const float* __restrict__ de_b1,
                                   const float* __restrict__ de_w2, const float* __restrict__ de_b2,
                                   const float* __restrict__ temp_sens,
                                   float* __restrict__ hnew,
                                   float* __restrict__ out_t) {
    int n = blockIdx.x * blockDim.x + threadIdx.x;
    if (n >= NN) return;
    const float* x = xt + (size_t)n * FF;

    float env[5];
#pragma unroll
    for (int i = 0; i < 5; ++i) env[i] = x[11 + i];
    const float tempv = env[0];

    float lc[3];
#pragma unroll
    for (int o = 0; o < 3; ++o) lc[o] = lice_cur[(size_t)n * LDIM + o];

    // env encoder: 5 -> 64 -> 64, plus pressure
    float t1[64];
    for (int o = 0; o < 64; ++o) {
        float s = en_b1[o];
#pragma unroll
        for (int i = 0; i < 5; ++i) s += env[i] * en_w1[o * 5 + i];
        t1[o] = fmaxf(s, 0.f);
    }
    float din[72];
    const float* pr = pressure + (size_t)n * HH;
    for (int o = 0; o < 64; ++o) {
        float s = en_b2[o];
        for (int i = 0; i < 64; ++i) s += t1[i] * en_w2[o * 64 + i];
        din[o] = s + pr[o];
    }
#pragma unroll
    for (int i = 0; i < 5; ++i) din[64 + i] = env[i];
#pragma unroll
    for (int o = 0; o < 3; ++o) din[69 + o] = lc[o];

    float h[64];
    for (int k = 0; k < 64; ++k) h[k] = hbuf[(size_t)n * HH + k];

    // GRU
    float hn[64];
    for (int j = 0; j < 64; ++j) {
        float ir = bih[j], iz = bih[64 + j], inn = bih[128 + j];
        const float* wr = wih + (size_t)j * 72;
        const float* wz = wih + (size_t)(64 + j) * 72;
        const float* wn = wih + (size_t)(128 + j) * 72;
        for (int k = 0; k < 72; ++k) {
            const float d = din[k];
            ir += d * wr[k]; iz += d * wz[k]; inn += d * wn[k];
        }
        float hr = bhh[j], hz = bhh[64 + j], hnn = bhh[128 + j];
        const float* vr = whh + (size_t)j * 64;
        const float* vz = whh + (size_t)(64 + j) * 64;
        const float* vn = whh + (size_t)(128 + j) * 64;
        for (int k = 0; k < 64; ++k) {
            const float hk = h[k];
            hr += hk * vr[k]; hz += hk * vz[k]; hnn += hk * vn[k];
        }
        const float r = 1.f / (1.f + __expf(-(ir + hr)));
        const float z = 1.f / (1.f + __expf(-(iz + hz)));
        const float nn2 = tanhf(inn + r * hnn);
        hn[j] = (1.f - z) * nn2 + z * h[j];
    }
    for (int j = 0; j < 64; ++j) hnew[(size_t)n * HH + j] = hn[j];

    // fecundity: 1 -> 32 -> 32 -> 1, softplus
    const float tn = (tempv - 10.f) * 0.2f;
    float a1[32];
#pragma unroll
    for (int i = 0; i < 32; ++i) a1[i] = fmaxf(tn * fe_w1[i] + fe_b1[i], 0.f);
    float b1a[32];
    for (int o = 0; o < 32; ++o) {
        float s = fe_b2[o];
#pragma unroll
        for (int i = 0; i < 32; ++i) s += a1[i] * fe_w2[o * 32 + i];
        b1a[o] = fmaxf(s, 0.f);
    }
    float cval = fe_b3[0];
#pragma unroll
    for (int i = 0; i < 32; ++i) cval += b1a[i] * fe_w3[i];
    const float fec = softplusf(cval);

    // decoder: 64 -> 64 -> 3, softplus, temp scaling
    float d1[64];
    for (int o = 0; o < 64; ++o) {
        float s = de_b1[o];
        for (int k = 0; k < 64; ++k) s += hn[k] * de_w1[o * 64 + k];
        d1[o] = fmaxf(s, 0.f);
    }
    const float scale = 1.f + temp_sens[0] * (fec - 1.f);
#pragma unroll
    for (int o = 0; o < 3; ++o) {
        float s = de_b2[o];
        for (int k = 0; k < 64; ++k) s += d1[k] * de_w2[o * 64 + k];
        out_t[(size_t)n * LDIM + o] = softplusf(s) * scale;
    }
}

// ------------------------------ launcher -----------------------------------

extern "C" void kernel_launch(void* const* d_in, const int* in_sizes, int n_in,
                              void* d_out, int out_size, void* d_ws, size_t ws_size,
                              hipStream_t stream) {
    const float* X        = (const float*)d_in[0];
    const float* lice0    = (const float*)d_in[1];
    const float* edist    = (const float*)d_in[2];
    const float* edir     = (const float*)d_in[3];
    const float* ee_w1    = (const float*)d_in[4];
    const float* ee_b1    = (const float*)d_in[5];
    const float* ee_w2    = (const float*)d_in[6];
    const float* ee_b2    = (const float*)d_in[7];
    const float* at_w1    = (const float*)d_in[8];
    const float* at_b1    = (const float*)d_in[9];
    const float* at_w2    = (const float*)d_in[10];
    const float* at_b2    = (const float*)d_in[11];
    const float* en_w1    = (const float*)d_in[12];
    const float* en_b1    = (const float*)d_in[13];
    const float* en_w2    = (const float*)d_in[14];
    const float* en_b2    = (const float*)d_in[15];
    const float* fe_w1    = (const float*)d_in[16];
    const float* fe_b1    = (const float*)d_in[17];
    const float* fe_w2    = (const float*)d_in[18];
    const float* fe_b2    = (const float*)d_in[19];
    const float* fe_w3    = (const float*)d_in[20];
    const float* fe_b3    = (const float*)d_in[21];
    const float* gru_wih  = (const float*)d_in[22];
    const float* gru_whh  = (const float*)d_in[23];
    const float* gru_bih  = (const float*)d_in[24];
    const float* gru_bhh  = (const float*)d_in[25];
    const float* de_w1    = (const float*)d_in[26];
    const float* de_b1    = (const float*)d_in[27];
    const float* de_w2    = (const float*)d_in[28];
    const float* de_b2    = (const float*)d_in[29];
    const float* log_beta = (const float*)d_in[30];
    const float* tsens    = (const float*)d_in[31];
    const int*   eidx     = (const int*)d_in[32];

    float* ws = (float*)d_ws;
    float*    hA       = ws;                          // N*H
    float*    hB       = hA + (size_t)NN * HH;        // N*H
    float*    pressure = hB + (size_t)NN * HH;        // N*H
    float*    asum     = pressure + (size_t)NN * HH;  // N
    unsigned* amax_u   = (unsigned*)(asum + NN);      // N
    float*    logits   = (float*)(amax_u + NN);       // E
    float*    aexp     = logits + EE;                 // E

    // h0 = zeros
    zero_f_kernel<<<(NN * HH + 255) / 256, 256, 0, stream>>>(hA, NN * HH);

    float* hcur = hA;
    float* hnew = hB;
    const float* licecur = lice0;
    float* out = (float*)d_out;

    const int nTiles = EE / 16;        // 20000
    const int edgeGrid = 1000;         // 2000 waves -> 10 uniform tiles/wave

    for (int t = 0; t < TT; ++t) {
        const float* xt = X + (size_t)t * NN * FF;
        float* out_t = out + (size_t)t * NN * LDIM;

        init_step_kernel<<<(NN * HH + 255) / 256, 256, 0, stream>>>(pressure, amax_u, asum);

        edge_logits_kernel<<<edgeGrid, 64, 0, stream>>>(
            xt, hcur, edist, edir, eidx,
            ee_w1, ee_b1, ee_w2, ee_b2,
            at_w1, at_b1, at_w2, at_b2,
            logits, amax_u, nTiles);

        edge_softmax_kernel<<<(EE + 255) / 256, 256, 0, stream>>>(
            logits, eidx, amax_u, aexp, asum);

        pressure_kernel<<<(EE * 4 + 255) / 256, 256, 0, stream>>>(
            aexp, asum, licecur, hcur, eidx, log_beta, pressure);

        node_update_kernel<<<(NN + 127) / 128, 128, 0, stream>>>(
            xt, hcur, pressure, licecur,
            en_w1, en_b1, en_w2, en_b2,
            fe_w1, fe_b1, fe_w2, fe_b2, fe_w3, fe_b3,
            gru_wih, gru_whh, gru_bih, gru_bhh,
            de_w1, de_b1, de_w2, de_b2,
            tsens, hnew, out_t);

        licecur = out_t;                 // lice feeds next step from d_out
        float* tmp = hcur; hcur = hnew; hnew = tmp;
    }
}
